// AblationFusionModel_54400055771771
// MI455X (gfx1250) — matrix operand order repaired and checked by reading, not compiled, not run on hardware
//
#include <hip/hip_runtime.h>
#include <hip/hip_bf16.h>
#include <math.h>

// ---------------------------------------------------------------------------
// Types
// ---------------------------------------------------------------------------
typedef __bf16 v16bf __attribute__((ext_vector_type(16)));
typedef float  v8f   __attribute__((ext_vector_type(8)));
typedef unsigned int u32x4 __attribute__((ext_vector_type(4)));
typedef int          i32x4 __attribute__((ext_vector_type(4)));
typedef int          i32x8 __attribute__((ext_vector_type(8)));

#define KDIM    512
#define HID     256
#define MTILE   64                 // rows of X per block
#define NTPW    2                  // 16-wide N tiles per wave (8 waves * 2 * 16 = 256)
#define KSTEP   32
#define NKSTEPS (KDIM / KSTEP)     // 16
#define KHALF   256                // floats per TDM half-tile row
#define LROW    260                // padded LDS row stride in dwords (TDM pad: 4dw / 256dw)
#define HALF_FLTS (MTILE * LROW)   // 16640 floats per half buffer

// workspace layout (bytes)
#define WS_W_OFF     0             // packed bf16 W1 fragments: 16*16*32*16*2 = 262144
#define WS_S_OFF     262144        // s[N] fp32
#define WS_VAGG_OFF  663552        // v_agg[512] fp32
#define WS_SCAL_OFF  665600        // [0]=max, [1]=sumExp

#if __has_builtin(__builtin_amdgcn_tensor_load_to_lds) && __has_builtin(__builtin_amdgcn_s_wait_tensorcnt)
#define HAVE_TDM 1
#else
#define HAVE_TDM 0
#endif

// ---------------------------------------------------------------------------
// Kernel 1: pack mil_w1 (fp32 [256,512]) into bf16 WMMA B-fragment layout.
// frag index = ((ntile*NKSTEPS + kstep)*32 + lane), 16 bf16 per lane.
// B (32x16, KxN): lane&15 = column, element e -> k = kstep*32 + (lane>>4)*8 + (e<8 ? e : e+8)
// ---------------------------------------------------------------------------
__global__ void pack_w_kernel(const float* __restrict__ w1, void* ws) {
    __bf16* wsW = (__bf16*)((char*)ws + WS_W_OFF);
    int tid = blockIdx.x * blockDim.x + threadIdx.x;          // 0..8191
    if (tid >= 16 * NKSTEPS * 32) return;
    int nt  = tid / (NKSTEPS * 32);
    int rem = tid % (NKSTEPS * 32);
    int ks  = rem / 32;
    int l   = rem % 32;
    int col = nt * 16 + (l & 15);
    const float* src = w1 + (size_t)col * KDIM + ks * KSTEP + (l >> 4) * 8;
    __bf16* dst = wsW + (size_t)tid * 16;
#pragma unroll
    for (int e = 0; e < 16; ++e) dst[e] = (__bf16)src[e < 8 ? e : e + 8];
}

// ---------------------------------------------------------------------------
// TDM: issue one 2-D tensor_load_to_lds for a 64-row x 256-float half tile,
// padded by the TDM (4 dwords after every 256) -> LDS row stride 260 dwords.
// ---------------------------------------------------------------------------
#if HAVE_TDM
__device__ inline void tdm_load_half(const float* X, int row0, int rows,
                                     int half, unsigned lds_byte_off) {
    unsigned long long ga =
        (unsigned long long)(const void*)(X + (size_t)row0 * KDIM + half * KHALF);
    u32x4 g0 = {
        1u,                                            // count=1, no gather
        lds_byte_off,                                  // lds_addr [63:32]
        (unsigned)(ga & 0xFFFFFFFFu),                  // global_addr lo
        (unsigned)((ga >> 32) & 0x01FFFFFFu) | (2u << 30) // addr hi | type=2
    };
    i32x8 g1 = {
        (int)((2u << 16) | (1u << 20) | (7u << 22) | (3u << 25)),
        //  data_size=4B | pad_enable | pad_interval=256dw | pad_amount=4dw
        (int)((unsigned)KHALF << 16),                  // tensor_dim0=256 -> [79:64... low16 in 31:16]
        (int)(((unsigned)rows & 0xFFFFu) << 16),       // tensor_dim1 low16
        (int)((unsigned)KHALF << 16),                  // tile_dim0=256 in [127:112]
        (int)MTILE,                                    // tile_dim1=64, tile_dim2=0
        (int)KDIM,                                     // tensor_dim0_stride=512 (lo32)
        0, 0                                           // stride hi / dim1_stride
    };
    i32x4 z4 = {0, 0, 0, 0};
#if defined(__clang_major__) && (__clang_major__ >= 23)
    i32x8 z8 = {0, 0, 0, 0, 0, 0, 0, 0};
    __builtin_amdgcn_tensor_load_to_lds(g0, g1, z4, z4, z8, 0);
#else
    __builtin_amdgcn_tensor_load_to_lds(g0, g1, z4, z4, 0);
#endif
}
#endif

// ---------------------------------------------------------------------------
// Kernel 2: s[i] = tanh(X[i,:] @ W1^T + b1) . w2 + b2
// TDM stages X tiles into LDS (double-buffered halves); WMMA bf16 compute.
// ---------------------------------------------------------------------------
__global__ __launch_bounds__(256) void mil_score_kernel(
    const float* __restrict__ X, const float* __restrict__ b1,
    const float* __restrict__ w2, const float* __restrict__ b2,
    void* ws, int N)
{
    const __bf16* wsW = (const __bf16*)((char*)ws + WS_W_OFF);
    float* sArr = (float*)((char*)ws + WS_S_OFF);

    __shared__ float xtile[2 * HALF_FLTS];   // 2 x 66560 B (TDM-padded rows)
    __shared__ float sloc[MTILE];

    const int wid  = threadIdx.x >> 5;
    const int lane = threadIdx.x & 31;
    const int r    = lane & 15;
    const int hi   = lane >> 4;
    const int row0 = blockIdx.x * MTILE;
    const int nt0  = wid * NTPW;
    const int rows = min(MTILE, N - row0);

#if HAVE_TDM
    if (wid == 0) {
        tdm_load_half(X, row0, rows, 0, (unsigned)(size_t)(const void*)&xtile[0]);
        tdm_load_half(X, row0, rows, 1, (unsigned)(size_t)(const void*)&xtile[HALF_FLTS]);
    }
#endif

    if (threadIdx.x < MTILE) sloc[threadIdx.x] = b2[0];

    const v8f vzero = {0.f, 0.f, 0.f, 0.f, 0.f, 0.f, 0.f, 0.f};
    v8f acc[4][NTPW];
#pragma unroll
    for (int m = 0; m < 4; ++m)
#pragma unroll
        for (int j = 0; j < NTPW; ++j) acc[m][j] = vzero;

    for (int half = 0; half < 2; ++half) {
        const float* buf = &xtile[half * HALF_FLTS];
#if HAVE_TDM
        if (wid == 0) {
            if (half == 0) __builtin_amdgcn_s_wait_tensorcnt(1);
            else           __builtin_amdgcn_s_wait_tensorcnt(0);
        }
        __syncthreads();
#else
        // fallback: cooperative staged copy (same padded layout)
        __syncthreads();
        for (int i = threadIdx.x; i < MTILE * (KHALF / 4); i += 256) {
            int rr = i >> 6, q = i & 63;
            int grow = row0 + rr; if (grow >= N) grow = N - 1;
            float4 v = *(const float4*)(X + (size_t)grow * KDIM + half * KHALF + q * 4);
            *(float4*)&xtile[half * HALF_FLTS + rr * LROW + q * 4] = v;
        }
        __syncthreads();
#endif
        // per-lane row bases inside the padded LDS tile
        const float* rowb[4];
#pragma unroll
        for (int m = 0; m < 4; ++m) rowb[m] = buf + (m * 16 + r) * LROW;

        for (int ksl = 0; ksl < NKSTEPS / 2; ++ksl) {
            const int off = ksl * KSTEP + hi * 8;        // dwords into padded row
            v16bf a[4];
#pragma unroll
            for (int m = 0; m < 4; ++m) {
                const float4* p1 = (const float4*)(rowb[m] + off);
                const float4* p2 = (const float4*)(rowb[m] + off + 16);
                float4 f0 = p1[0], f1 = p1[1], f2 = p2[0], f3 = p2[1];
                v16bf av;
                av[0]  = (__bf16)f0.x; av[1]  = (__bf16)f0.y; av[2]  = (__bf16)f0.z; av[3]  = (__bf16)f0.w;
                av[4]  = (__bf16)f1.x; av[5]  = (__bf16)f1.y; av[6]  = (__bf16)f1.z; av[7]  = (__bf16)f1.w;
                av[8]  = (__bf16)f2.x; av[9]  = (__bf16)f2.y; av[10] = (__bf16)f2.z; av[11] = (__bf16)f2.w;
                av[12] = (__bf16)f3.x; av[13] = (__bf16)f3.y; av[14] = (__bf16)f3.z; av[15] = (__bf16)f3.w;
                a[m] = av;
            }
            const int ks = half * (NKSTEPS / 2) + ksl;   // global k-step for B frags
#pragma unroll
            for (int j = 0; j < NTPW; ++j) {
                const int nt = nt0 + j;
                const v16bf bfrag =
                    *(const v16bf*)(wsW + (((size_t)nt * NKSTEPS + ks) * 32 + lane) * 16);
#pragma unroll
                for (int m = 0; m < 4; ++m) {
                    acc[m][j] = __builtin_amdgcn_wmma_f32_16x16x32_bf16(
                        false, a[m], false, bfrag, (short)0, acc[m][j], false, false);
                }
            }
        }
    }

    // epilogue: tanh(h + b1) * w2, reduce 16 columns across lanes, ds-atomic
#pragma unroll
    for (int j = 0; j < NTPW; ++j) {
        const int col = (nt0 + j) * 16 + r;
        const float bb = b1[col];
        const float ww = w2[col];
#pragma unroll
        for (int m = 0; m < 4; ++m) {
#pragma unroll
            for (int jj = 0; jj < 8; ++jj) {
                float val = tanhf(acc[m][j][jj] + bb) * ww;
                val += __shfl_xor(val, 8, 32);
                val += __shfl_xor(val, 4, 32);
                val += __shfl_xor(val, 2, 32);
                val += __shfl_xor(val, 1, 32);
                if (r == 0) atomicAdd(&sloc[m * 16 + jj + 8 * hi], val);
            }
        }
    }
    __syncthreads();
    if (threadIdx.x < MTILE) {
        const int row = row0 + threadIdx.x;
        if (row < N) sArr[row] = sloc[threadIdx.x];
    }
}

// ---------------------------------------------------------------------------
// Kernel 3: softmax stats (max, sum of exp) over s[0..N); zero v_agg.
// ---------------------------------------------------------------------------
__global__ __launch_bounds__(1024) void softmax_stats_kernel(void* ws, int N) {
    float* sArr = (float*)((char*)ws + WS_S_OFF);
    float* vagg = (float*)((char*)ws + WS_VAGG_OFF);
    float* scal = (float*)((char*)ws + WS_SCAL_OFF);
    __shared__ float red[1024];
    const int t = threadIdx.x;

    float m = -1e30f;
    for (int i = t; i < N; i += 1024) m = fmaxf(m, sArr[i]);
    red[t] = m; __syncthreads();
    for (int s = 512; s > 0; s >>= 1) {
        if (t < s) red[t] = fmaxf(red[t], red[t + s]);
        __syncthreads();
    }
    const float mx = red[0];
    __syncthreads();

    float sum = 0.f;
    for (int i = t; i < N; i += 1024) sum += expf(sArr[i] - mx);
    red[t] = sum; __syncthreads();
    for (int s = 512; s > 0; s >>= 1) {
        if (t < s) red[t] += red[t + s];
        __syncthreads();
    }
    if (t == 0) { scal[0] = mx; scal[1] = red[0]; }
    if (t < 512) vagg[t] = 0.f;
}

// ---------------------------------------------------------------------------
// Kernel 4: v_agg[c] = sum_i softmax(s)_i * X[i,c]   (coalesced stream of X)
// ---------------------------------------------------------------------------
__global__ __launch_bounds__(512) void aggregate_kernel(
    const float* __restrict__ X, void* ws, int N)
{
    const float* sArr = (const float*)((char*)ws + WS_S_OFF);
    float* vagg = (float*)((char*)ws + WS_VAGG_OFF);
    const float* scal = (const float*)((char*)ws + WS_SCAL_OFF);

    const int c = threadIdx.x;
    const float mx = scal[0];
    const float invZ = 1.f / scal[1];
    const int per = (N + gridDim.x - 1) / gridDim.x;
    const int i0 = blockIdx.x * per;
    const int i1 = min(i0 + per, N);

    float acc = 0.f;
    for (int i = i0; i < i1; ++i) {
        const float w = expf(sArr[i] - mx) * invZ;
        acc += w * X[(size_t)i * KDIM + c];
    }
    atomicAdd(&vagg[c], acc);
}

// ---------------------------------------------------------------------------
// Kernel 5: fusion tail, one 64-thread block.
// ---------------------------------------------------------------------------
__device__ inline float gelu_exact(float x) {
    return 0.5f * x * (1.f + erff(x * 0.70710678118654752f));
}

__device__ inline void ln_inplace(float* sh, int n, const float* g, const float* b, int t) {
    __syncthreads();
    float mean = 0.f;
    for (int i = 0; i < n; ++i) mean += sh[i];
    mean /= (float)n;
    float var = 0.f;
    for (int i = 0; i < n; ++i) { float d = sh[i] - mean; var += d * d; }
    var /= (float)n;
    const float inv = rsqrtf(var + 1e-5f);
    float v = 0.f;
    if (t < n) v = (sh[t] - mean) * inv * g[t] + b[t];
    __syncthreads();
    if (t < n) sh[t] = v;
    __syncthreads();
}

// GCA on singleton sequences: softmax over a single key == 1.0 exactly,
// so attn == v-projection; q,k are mathematically dead and skipped.
__device__ inline void gca(const float* xq, const float* xkv,
                           const float* qkv_w, const float* qkv_b,
                           const float* out_w, const float* out_b,
                           const float* gate_w, const float* gate_b,
                           const float* ln_g, const float* ln_b,
                           float* vp, float* ao, float* dst, int t)
{
    float a = qkv_b[128 + t];
    for (int k = 0; k < 64; ++k) a += qkv_w[(128 + t) * 64 + k] * xkv[k];
    vp[t] = a; __syncthreads();

    float o = out_b[t];
    for (int k = 0; k < 64; ++k) o += out_w[t * 64 + k] * vp[k];
    ao[t] = o; __syncthreads();

    float gz = gate_b[t];
    for (int k = 0; k < 64; ++k) gz += gate_w[t * 128 + k] * xq[k];
    for (int k = 0; k < 64; ++k) gz += gate_w[t * 128 + 64 + k] * ao[k];
    const float g = 1.f / (1.f + expf(-gz));
    const float outv = g * ao[t] + (1.f - g) * xq[t];
    __syncthreads();
    dst[t] = outv;
    ln_inplace(dst, 64, ln_g, ln_b, t);
}

__global__ __launch_bounds__(64) void tail_kernel(
    const float* __restrict__ clin,
    const float* visp_w, const float* visp_b, const float* visp_g, const float* visp_be,
    const float* clinp_w, const float* clinp_b, const float* clinp_g, const float* clinp_be,
    const float* cqv_qkv_w, const float* cqv_qkv_b, const float* cqv_out_w, const float* cqv_out_b,
    const float* cqv_gate_w, const float* cqv_gate_b, const float* cqv_ln_g, const float* cqv_ln_b,
    const float* vqc_qkv_w, const float* vqc_qkv_b, const float* vqc_out_w, const float* vqc_out_b,
    const float* vqc_gate_w, const float* vqc_gate_b, const float* vqc_ln_g, const float* vqc_ln_b,
    const float* cls_w1, const float* cls_b1, const float* cls_g, const float* cls_be,
    const float* cls_w2, const float* cls_b2,
    void* ws, float* __restrict__ out)
{
    const float* vagg = (const float*)((char*)ws + WS_VAGG_OFF);
    __shared__ float sv[64], sc[64], vp[64], ao[64], cf[64], vf[64], h32[32];
    const int t = threadIdx.x;

    float acc = visp_b[t];
    for (int k = 0; k < 512; ++k) acc += visp_w[t * 512 + k] * vagg[k];
    sv[t] = acc;
    ln_inplace(sv, 64, visp_g, visp_be, t);
    sv[t] = gelu_exact(sv[t]); __syncthreads();

    acc = clinp_b[t];
    for (int k = 0; k < 128; ++k) acc += clinp_w[t * 128 + k] * clin[k];
    sc[t] = acc;
    ln_inplace(sc, 64, clinp_g, clinp_be, t);
    sc[t] = gelu_exact(sc[t]); __syncthreads();

    gca(sc, sv, cqv_qkv_w, cqv_qkv_b, cqv_out_w, cqv_out_b,
        cqv_gate_w, cqv_gate_b, cqv_ln_g, cqv_ln_b, vp, ao, cf, t);
    gca(sv, sc, vqc_qkv_w, vqc_qkv_b, vqc_out_w, vqc_out_b,
        vqc_gate_w, vqc_gate_b, vqc_ln_g, vqc_ln_b, vp, ao, vf, t);

    if (t < 32) {
        float a = cls_b1[t];
        for (int k = 0; k < 64; ++k) a += cls_w1[t * 128 + k] * cf[k];
        for (int k = 0; k < 64; ++k) a += cls_w1[t * 128 + 64 + k] * vf[k];
        h32[t] = a;
    }
    ln_inplace(h32, 32, cls_g, cls_be, t);
    if (t < 32) h32[t] = gelu_exact(h32[t]);
    __syncthreads();
    if (t == 0) {
        float a = cls_b2[0];
        for (int k = 0; k < 32; ++k) a += cls_w2[k] * h32[k];
        out[0] = a;
    }
}

// ---------------------------------------------------------------------------
// Launch
// ---------------------------------------------------------------------------
extern "C" void kernel_launch(void* const* d_in, const int* in_sizes, int n_in,
                              void* d_out, int out_size, void* d_ws, size_t ws_size,
                              hipStream_t stream) {
    const float* vis_bag   = (const float*)d_in[0];
    const float* clin      = (const float*)d_in[1];
    const float* mil_w1    = (const float*)d_in[2];
    const float* mil_b1    = (const float*)d_in[3];
    const float* mil_w2    = (const float*)d_in[4];
    const float* mil_b2    = (const float*)d_in[5];
    const float* visp_w    = (const float*)d_in[6];
    const float* visp_b    = (const float*)d_in[7];
    const float* visp_g    = (const float*)d_in[8];
    const float* visp_be   = (const float*)d_in[9];
    const float* clinp_w   = (const float*)d_in[10];
    const float* clinp_b   = (const float*)d_in[11];
    const float* clinp_g   = (const float*)d_in[12];
    const float* clinp_be  = (const float*)d_in[13];
    const float* cqv_qkv_w = (const float*)d_in[14];
    const float* cqv_qkv_b = (const float*)d_in[15];
    const float* cqv_out_w = (const float*)d_in[16];
    const float* cqv_out_b = (const float*)d_in[17];
    const float* cqv_gate_w= (const float*)d_in[18];
    const float* cqv_gate_b= (const float*)d_in[19];
    const float* cqv_ln_g  = (const float*)d_in[20];
    const float* cqv_ln_b  = (const float*)d_in[21];
    const float* vqc_qkv_w = (const float*)d_in[22];
    const float* vqc_qkv_b = (const float*)d_in[23];
    const float* vqc_out_w = (const float*)d_in[24];
    const float* vqc_out_b = (const float*)d_in[25];
    const float* vqc_gate_w= (const float*)d_in[26];
    const float* vqc_gate_b= (const float*)d_in[27];
    const float* vqc_ln_g  = (const float*)d_in[28];
    const float* vqc_ln_b  = (const float*)d_in[29];
    const float* cls_w1    = (const float*)d_in[30];
    const float* cls_b1    = (const float*)d_in[31];
    const float* cls_g     = (const float*)d_in[32];
    const float* cls_be    = (const float*)d_in[33];
    const float* cls_w2    = (const float*)d_in[34];
    const float* cls_b2    = (const float*)d_in[35];

    const int N = in_sizes[0] / KDIM;   // 100000

    pack_w_kernel<<<32, 256, 0, stream>>>(mil_w1, d_ws);

    const int nblk = (N + MTILE - 1) / MTILE;
    mil_score_kernel<<<nblk, 256, 0, stream>>>(vis_bag, mil_b1, mil_w2, mil_b2, d_ws, N);

    softmax_stats_kernel<<<1, 1024, 0, stream>>>(d_ws, N);

    aggregate_kernel<<<256, 512, 0, stream>>>(vis_bag, d_ws, N);

    tail_kernel<<<1, 64, 0, stream>>>(clin,
        visp_w, visp_b, visp_g, visp_be,
        clinp_w, clinp_b, clinp_g, clinp_be,
        cqv_qkv_w, cqv_qkv_b, cqv_out_w, cqv_out_b,
        cqv_gate_w, cqv_gate_b, cqv_ln_g, cqv_ln_b,
        vqc_qkv_w, vqc_qkv_b, vqc_out_w, vqc_out_b,
        vqc_gate_w, vqc_gate_b, vqc_ln_g, vqc_ln_b,
        cls_w1, cls_b1, cls_g, cls_be, cls_w2, cls_b2,
        d_ws, (float*)d_out);
}
